// MLP_77859167141930
// MI455X (gfx1250) — matrix operand order repaired and checked by reading, not compiled, not run on hardware
//
#include <hip/hip_runtime.h>
#include <hip/hip_bf16.h>

// ---------------- WMMA types ----------------
typedef __attribute__((ext_vector_type(16))) __bf16 v16bf;
typedef __attribute__((ext_vector_type(2)))  __bf16 v2bf;
typedef __attribute__((ext_vector_type(2)))  float  v2f;
typedef __attribute__((ext_vector_type(8)))  float  v8f;

union FragBF { unsigned short u[16]; v16bf v; };

#define BLK_M 128
#define BLK_N 128
#define BLK_K 32
#define LDS_PITCH (BLK_K + 8)   // ushort pitch; keeps 16B alignment for ds_load_b128

// f32x2 -> packed bf16x2 via hardware cvt (RNE), no software rounding sequence
__device__ __forceinline__ unsigned pack2bf(float lo, float hi) {
    v2f f; f[0] = lo; f[1] = hi;
    union { v2bf b; unsigned u; } cv;
    cv.b = __builtin_convertvector(f, v2bf);
    return cv.u;
}

// ---------------- Tiled bf16 WMMA GEMM: C[M,N] = A[M,K] @ W[K,N] (+bias)(+relu) ----------------
// 256 threads = 8 waves; wave grid 4(M) x 2(N); each wave: 32x64 output = 2x4 WMMA tiles.
// Requires K % 32 == 0 (true for 512/256/128 here).
__global__ __launch_bounds__(256)
void gat_gemm_wmma(const float* __restrict__ A, const float* __restrict__ W,
                   const float* __restrict__ bias, float* __restrict__ C,
                   int M, int K, int N, int act)
{
    __shared__ unsigned short As[BLK_M][LDS_PITCH];   // A tile, bf16, [m][k]
    __shared__ unsigned short Bs[BLK_N][LDS_PITCH];   // B tile, bf16, transposed [n][k]

    const int tid  = threadIdx.x;
    const int wave = tid >> 5;
    const int lane = tid & 31;
    const int lh   = lane >> 4;
    const int ll   = lane & 15;

    const int m0 = blockIdx.x * BLK_M;
    const int n0 = blockIdx.y * BLK_N;
    const bool nfull = (n0 + BLK_N) <= N;

    const int waveM = (wave & 3) << 5;   // 0,32,64,96
    const int waveN = (wave >> 2) << 6;  // 0,64

    v8f acc[2][4];
    #pragma unroll
    for (int i = 0; i < 2; i++)
        #pragma unroll
        for (int t = 0; t < 4; t++)
            #pragma unroll
            for (int r = 0; r < 8; r++) acc[i][t][r] = 0.0f;

    // prefetch registers
    float aR[4][4];        // 4 float4 slots of the A tile
    float bR[2][2][4];     // 2 slots x {k, k+1} x float4 along n

    // ---- tile loaders (branch-free fast paths; row/col clamped, never OOB) ----
    auto loadA = [&](int kb) {
        #pragma unroll
        for (int i = 0; i < 4; i++) {
            int s   = i * 256 + tid;          // 1024 float4 slots: 128 rows x 8
            int row = s >> 3, k4 = s & 7;
            int gr  = m0 + row; gr = gr < M ? gr : M - 1;     // clamp (dup rows unused)
            *(float4*)aR[i] = *(const float4*)(A + (size_t)gr * K + kb + k4 * 4);
        }
    };
    auto loadB = [&](int kb) {
        #pragma unroll
        for (int i = 0; i < 2; i++) {
            int s  = i * 256 + tid;           // 512 pair-slots: 16 kp x 32 n4
            int kp = s >> 5, n4 = s & 31;
            int gk = kb + 2 * kp;
            if (nfull) {
                const float* wp = W + (size_t)gk * N + n0 + n4 * 4;
                *(float4*)bR[i][0] = *(const float4*)wp;
                *(float4*)bR[i][1] = *(const float4*)(wp + N);
            } else {
                #pragma unroll
                for (int j = 0; j < 4; j++) {
                    int gc = n0 + n4 * 4 + j; gc = gc < N ? gc : N - 1;  // clamp
                    bR[i][0][j] = W[(size_t)gk * N + gc];
                    bR[i][1][j] = W[(size_t)(gk + 1) * N + gc];
                }
            }
        }
    };
    auto storeTiles = [&]() {
        #pragma unroll
        for (int i = 0; i < 4; i++) {
            int s = i * 256 + tid;
            int row = s >> 3, k4 = s & 7;
            uint2 pk;
            pk.x = pack2bf(aR[i][0], aR[i][1]);
            pk.y = pack2bf(aR[i][2], aR[i][3]);
            *(uint2*)&As[row][k4 * 4] = pk;   // 8B aligned: 80*row + 8*k4
        }
        #pragma unroll
        for (int i = 0; i < 2; i++) {
            int s = i * 256 + tid;
            int kp = s >> 5, n4 = s & 31;
            #pragma unroll
            for (int j = 0; j < 4; j++)
                *(unsigned*)&Bs[n4 * 4 + j][2 * kp] = pack2bf(bR[i][0][j], bR[i][1][j]);
        }
    };
    auto compute = [&]() {
        FragBF af[2];
        #pragma unroll
        for (int ii = 0; ii < 2; ii++) {
            const int arow  = waveM + ii * 16 + ll;
            const int abase = lh * 8;         // lane<16: K{0..7,16..23}; else K{8..15,24..31}
            #pragma unroll
            for (int i = 0; i < 16; i++)
                af[ii].u[i] = As[arow][abase + (i < 8 ? i : i + 8)];
        }
        #pragma unroll
        for (int t = 0; t < 4; t++) {
            FragBF bf;
            const int bcol  = waveN + t * 16 + ll;
            const int bbase = lh * 8;         // lane<16: K{0..7,16..23}; else K{8..15,24..31}
            #pragma unroll
            for (int i = 0; i < 16; i++) bf.u[i] = Bs[bcol][bbase + (i < 8 ? i : i + 8)];
            acc[0][t] = __builtin_amdgcn_wmma_f32_16x16x32_bf16(
                false, af[0].v, false, bf.v, (short)0, acc[0][t], false, false);
            acc[1][t] = __builtin_amdgcn_wmma_f32_16x16x32_bf16(
                false, af[1].v, false, bf.v, (short)0, acc[1][t], false, false);
        }
    };

    // ---- software-pipelined main loop ----
    loadA(0); loadB(0);
    storeTiles();
    __syncthreads();
    int kb = 0;
    for (;;) {
        const int kn = kb + BLK_K;
        const bool more = kn < K;
        if (more) { loadA(kn); loadB(kn); }   // global loads in flight during WMMAs
        compute();
        if (!more) break;
        __syncthreads();
        storeTiles();
        __syncthreads();
        kb = kn;
    }

    // ---- epilogue: D element r -> row = r + 8*lh, col = ll ----
    #pragma unroll
    for (int i = 0; i < 2; i++) {
        #pragma unroll
        for (int t = 0; t < 4; t++) {
            int col = n0 + waveN + t * 16 + ll;
            if (col >= N) continue;
            float bv = bias ? bias[col] : 0.0f;
            #pragma unroll
            for (int r = 0; r < 8; r++) {
                int row = m0 + waveM + i * 16 + r + 8 * lh;
                if (row >= M) continue;
                float v = acc[i][t][r] + bv;
                if (act) v = v > 0.0f ? v : 0.0f;
                C[(size_t)row * N + col] = v;
            }
        }
    }
}

// ---------------- attention score dot products: s_src = h@a_src, s_dst = h@a_dst ----------------
__global__ void gat_scores(const float* __restrict__ h,
                           const float* __restrict__ a_src, const float* __restrict__ a_dst,
                           float* __restrict__ ssrc, float* __restrict__ sdst,
                           int n, int F)
{
    int node = blockIdx.x * (blockDim.x >> 5) + (threadIdx.x >> 5);
    int lane = threadIdx.x & 31;
    if (node >= n) return;
    const float* row = h + (size_t)node * F;
    float s1 = 0.0f, s2 = 0.0f;
    for (int f = lane; f < F; f += 32) {
        float v = row[f];
        s1 += v * a_src[f];
        s2 += v * a_dst[f];
    }
    #pragma unroll
    for (int o = 16; o > 0; o >>= 1) {
        s1 += __shfl_down(s1, o, 32);
        s2 += __shfl_down(s2, o, 32);
    }
    if (lane == 0) { ssrc[node] = s1; sdst[node] = s2; }
}

// ---------------- float <-> monotone uint key (for atomic max on floats) ----------------
__device__ __forceinline__ unsigned fkey(float f) {
    unsigned u = __float_as_uint(f);
    return (u & 0x80000000u) ? ~u : (u | 0x80000000u);
}
__device__ __forceinline__ float fkeyinv(unsigned k) {
    unsigned u = (k & 0x80000000u) ? (k & 0x7FFFFFFFu) : ~k;
    return __uint_as_float(u);
}

__device__ __forceinline__ void edge_sd(const long long* ei, int E, int e, int& s, int& d) {
    if (e < E) { s = (int)ei[e]; d = (int)ei[(size_t)E + e]; }
    else       { s = e - E; d = s; }   // appended self-loops
}

// ---------------- edge pass 1: e = leakyrelu(ssrc[s]+sdst[d]); segment max ----------------
__global__ void gat_edge_max(const long long* __restrict__ ei, int E, int n,
                             const float* __restrict__ ssrc, const float* __restrict__ sdst,
                             float* __restrict__ ebuf, unsigned* __restrict__ mkey)
{
    int e = blockIdx.x * blockDim.x + threadIdx.x;
    if (e >= E + n) return;
    int s, d; edge_sd(ei, E, e, s, d);
    float v = ssrc[s] + sdst[d];
    v = v > 0.0f ? v : 0.2f * v;
    ebuf[e] = v;
    atomicMax(&mkey[d], fkey(v));
}

// ---------------- edge pass 2: ex = exp(e - m[d]); segment sum ----------------
__global__ void gat_edge_exp(const long long* __restrict__ ei, int E, int n,
                             float* __restrict__ ebuf,
                             const unsigned* __restrict__ mkey, float* __restrict__ denom)
{
    int e = blockIdx.x * blockDim.x + threadIdx.x;
    if (e >= E + n) return;
    int s, d; edge_sd(ei, E, e, s, d);
    float m  = fkeyinv(mkey[d]);
    float ex = __expf(ebuf[e] - m);
    ebuf[e] = ex;
    atomicAdd(&denom[d], ex);
}

// ---------------- edge pass 3: agg[d] += alpha * h[s]; one wave per edge ----------------
__global__ void gat_edge_scatter(const long long* __restrict__ ei, int E, int n,
                                 const float* __restrict__ h, const float* __restrict__ ebuf,
                                 const float* __restrict__ denom, float* __restrict__ agg, int F)
{
    int e    = blockIdx.x * (blockDim.x >> 5) + (threadIdx.x >> 5);
    int lane = threadIdx.x & 31;
    if (e >= E + n) return;
    int s, d; edge_sd(ei, E, e, s, d);
    float alpha = ebuf[e] / (denom[d] + 1e-16f);
    const float4* hs = (const float4*)(h + (size_t)s * F);
    float* ad = agg + (size_t)d * F;
    for (int c = lane; c < (F >> 2); c += 32) {
        float4 v = hs[c];
        atomicAdd(ad + c * 4 + 0, alpha * v.x);
        atomicAdd(ad + c * 4 + 1, alpha * v.y);
        atomicAdd(ad + c * 4 + 2, alpha * v.z);
        atomicAdd(ad + c * 4 + 3, alpha * v.w);
    }
}

// ---------------- bias (+relu), in place ----------------
__global__ void gat_bias_act(float* __restrict__ x, const float* __restrict__ b,
                             int total, int F, int act)
{
    int i = blockIdx.x * blockDim.x + threadIdx.x;
    if (i >= total) return;
    float v = x[i] + b[i % F];
    if (act) v = v > 0.0f ? v : 0.0f;
    x[i] = v;
}

// ---------------- orchestration ----------------
extern "C" void kernel_launch(void* const* d_in, const int* in_sizes, int n_in,
                              void* d_out, int out_size, void* d_ws, size_t ws_size,
                              hipStream_t stream)
{
    const float*     x   = (const float*)d_in[0];
    const long long* ei  = (const long long*)d_in[1];
    const float*     W1  = (const float*)d_in[2];
    const float*     a1s = (const float*)d_in[3];
    const float*     a1d = (const float*)d_in[4];
    const float*     b1  = (const float*)d_in[5];
    const float*     W2  = (const float*)d_in[6];
    const float*     a2s = (const float*)d_in[7];
    const float*     a2d = (const float*)d_in[8];
    const float*     b2  = (const float*)d_in[9];
    const float*     Wl1 = (const float*)d_in[10];
    const float*     bl1 = (const float*)d_in[11];
    const float*     Wl2 = (const float*)d_in[12];
    const float*     bl2 = (const float*)d_in[13];

    const int H1   = in_sizes[3];            // 256
    const int Fin  = in_sizes[2] / H1;       // 512
    const int n    = in_sizes[0] / Fin;      // 50000
    const int H2   = in_sizes[7];            // 256
    const int H3   = in_sizes[11];           // 128
    const int Ccls = in_sizes[13];           // 186
    const int E    = in_sizes[1] / 2;        // 1.6M
    const int Etot = E + n;
    const int Hmax = H1 > H2 ? H1 : H2;

    // workspace carve (stream-ordered reuse: hbuf/aggbuf ping-pong across layers)
    char* p = (char*)d_ws;
    auto carve = [&](size_t bytes) -> void* {
        void* r = (void*)p;
        p += (bytes + 255) & ~(size_t)255;
        return r;
    };
    float*    hbuf   = (float*)carve((size_t)n * Hmax * sizeof(float));
    float*    aggbuf = (float*)carve((size_t)n * Hmax * sizeof(float));
    float*    ssrc   = (float*)carve((size_t)n * sizeof(float));
    float*    sdst   = (float*)carve((size_t)n * sizeof(float));
    float*    denom  = (float*)carve((size_t)n * sizeof(float));
    unsigned* mkey   = (unsigned*)carve((size_t)n * sizeof(unsigned));
    float*    ebuf   = (float*)carve((size_t)Etot * sizeof(float));

    auto gemm = [&](const float* A, const float* W, const float* bias, float* C,
                    int M, int K, int N, int act) {
        dim3 grid((M + BLK_M - 1) / BLK_M, (N + BLK_N - 1) / BLK_N);
        gat_gemm_wmma<<<grid, dim3(256), 0, stream>>>(A, W, bias, C, M, K, N, act);
    };

    auto gat_layer = [&](const float* in, int Fi, const float* W,
                         const float* as_, const float* ad_, const float* bb,
                         int H, float* hout, float* aggout) {
        gemm(in, W, nullptr, hout, n, Fi, H, 0);
        gat_scores<<<dim3((n + 7) / 8), dim3(256), 0, stream>>>(hout, as_, ad_, ssrc, sdst, n, H);
        hipMemsetAsync(mkey,  0, (size_t)n * sizeof(unsigned), stream);
        hipMemsetAsync(denom, 0, (size_t)n * sizeof(float),    stream);
        hipMemsetAsync(aggout, 0, (size_t)n * H * sizeof(float), stream);
        gat_edge_max    <<<dim3((Etot + 255) / 256), dim3(256), 0, stream>>>(ei, E, n, ssrc, sdst, ebuf, mkey);
        gat_edge_exp    <<<dim3((Etot + 255) / 256), dim3(256), 0, stream>>>(ei, E, n, ebuf, mkey, denom);
        gat_edge_scatter<<<dim3((Etot + 7) / 8),     dim3(256), 0, stream>>>(ei, E, n, hout, ebuf, denom, aggout, H);
        gat_bias_act    <<<dim3((n * H + 255) / 256), dim3(256), 0, stream>>>(aggout, bb, n * H, H, 1);
    };

    // layer 1: x -> hbuf -> aggbuf (act1)
    gat_layer(x, Fin, W1, a1s, a1d, b1, H1, hbuf, aggbuf);
    // layer 2: act1(aggbuf) -> hbuf (h2) -> aggbuf (act2; safe: gemm consumed act1 first)
    gat_layer(aggbuf, H1, W2, a2s, a2d, b2, H2, hbuf, aggbuf);
    // linear head: act2 -> hbuf (relu(x@Wl1+bl1)) -> d_out (x@Wl2+bl2)
    gemm(aggbuf, Wl1, bl1, hbuf, n, H2, H3, 1);
    gemm(hbuf, Wl2, bl2, (float*)d_out, n, H3, Ccls, 0);
}
